// SpikeDrivenSelfAttention_56384330662230
// MI455X (gfx1250) — 16-bit and 8-bit matrix operand order repaired and checked by reading, compiled, not run on hardware
//
#include <hip/hip_runtime.h>
#include <hip/hip_bf16.h>


#define DIMC    768
#define NHEADS  12
#define HDIM    64
#define BATCH   8
#define SEQ     1024
#define TSTEPS  4
#define DECAY   0.6065306597126334f
#define THRESH  1.0f

typedef __attribute__((ext_vector_type(2)))  float    v2f;
typedef __attribute__((ext_vector_type(8)))  float    v8f;
typedef __attribute__((ext_vector_type(2)))  int      v2i;
typedef __attribute__((ext_vector_type(4)))  int      v4i;
typedef __attribute__((ext_vector_type(8)))  int      v8i;
typedef __attribute__((ext_vector_type(8)))  _Float16 v8h;
typedef __attribute__((ext_vector_type(16))) _Float16 v16h;

__device__ __forceinline__ float lif_spike(float inp) {
    float v = 0.0f, agg = 0.0f;
#pragma unroll
    for (int t = 0; t < TSTEPS; ++t) {
        v = v * DECAY + inp;
        float sp = (v >= THRESH) ? 1.0f : 0.0f;
        v -= THRESH * sp;
        agg += sp;
    }
    return (agg > 0.0f) ? 1.0f : 0.0f;
}

__device__ __forceinline__ unsigned lds_offset(const void* p) {
    return (unsigned)(unsigned long long)p;
}

__global__ __launch_bounds__(256) void qkv_lif_kernel(
    const float* __restrict__ x,
    const float* __restrict__ Wq, const float* __restrict__ Wk,
    const float* __restrict__ Wv,
    const float* __restrict__ bq, const float* __restrict__ bk,
    const float* __restrict__ bv,
    unsigned char* __restrict__ sq, unsigned char* __restrict__ sk,
    unsigned char* __restrict__ sv)
{
    __shared__ float As[128 * 36];
    __shared__ float Bs[16 * 258];

    const int m0   = blockIdx.y * 128;
    const int n0   = blockIdx.x * 128;
    const int tid  = threadIdx.x;
    const int lane = tid & 31;
    const int wave = tid >> 5;
    const int mw   = wave >> 1;
    const int nw   = wave & 1;
    const int hi   = lane >> 4;
    const int lm   = lane & 15;

    const float* Wsel = (n0 < DIMC) ? Wq : (n0 < 2 * DIMC ? Wk : Wv);
    const int    obas = n0 % DIMC;

    v8f acc[2][4] = {};

    for (int k0 = 0; k0 < DIMC; k0 += 32) {
        __syncthreads();
        {
            int m  = tid >> 1;
            int kk = (tid & 1) * 16;
            const float* src = x + (size_t)(m0 + m) * DIMC + k0 + kk;
#pragma unroll
            for (int i = 0; i < 16; i += 4)
                *(float4*)(&As[m * 36 + kk + i]) = *(const float4*)(src + i);
        }
        {
            int oc = tid >> 1;
            int kk = (tid & 1) * 16;
            const float* src = Wsel + (size_t)(obas + oc) * DIMC + k0 + kk;
#pragma unroll
            for (int i = 0; i < 16; i += 4) {
                float4 w4 = *(const float4*)(src + i);
                int k = kk + i;
                Bs[((k + 0) >> 1) * 258 + oc * 2 + ((k + 0) & 1)] = w4.x;
                Bs[((k + 1) >> 1) * 258 + oc * 2 + ((k + 1) & 1)] = w4.y;
                Bs[((k + 2) >> 1) * 258 + oc * 2 + ((k + 2) & 1)] = w4.z;
                Bs[((k + 3) >> 1) * 258 + oc * 2 + ((k + 3) & 1)] = w4.w;
            }
        }
        __syncthreads();

        const float* Arow0 = &As[(mw * 32 + lm) * 36];
        const float* Arow1 = &As[(mw * 32 + 16 + lm) * 36];
#pragma unroll
        for (int ks = 0; ks < 32; ks += 4) {
            v2f a0 = *(const v2f*)(&Arow0[ks + 2 * hi]);
            v2f a1 = *(const v2f*)(&Arow1[ks + 2 * hi]);
#pragma unroll
            for (int s = 0; s < 4; ++s) {
                int ocol = nw * 64 + s * 16 + lm;
                v2f b = *(const v2f*)(&Bs[((ks >> 1) + hi) * 258 + ocol * 2]);
                acc[0][s] = __builtin_amdgcn_wmma_f32_16x16x4_f32(
                    false, a0, false, b, (short)0, acc[0][s], false, false);
                acc[1][s] = __builtin_amdgcn_wmma_f32_16x16x4_f32(
                    false, a1, false, b, (short)0, acc[1][s], false, false);
            }
        }
    }

#pragma unroll
    for (int s = 0; s < 4; ++s) {
        int o     = n0 + nw * 64 + s * 16 + lm;
        int which = o / DIMC;
        int c     = o - which * DIMC;
        const float*   bias = (which == 0) ? bq : (which == 1 ? bk : bv);
        unsigned char* sp   = (which == 0) ? sq : (which == 1 ? sk : sv);
        float bb = bias[c];
        int h = c >> 6, d = c & 63;
#pragma unroll
        for (int ms = 0; ms < 2; ++ms) {
#pragma unroll
            for (int g = 0; g < 8; ++g) {
                int m  = m0 + mw * 32 + ms * 16 + g + 8 * hi;
                int bi = m >> 10, n = m & 1023;
                float spk = lif_spike(acc[ms][s][g] + bb);
                sp[((size_t)(bi * NHEADS + h) * SEQ + n) * HDIM + d] =
                    (unsigned char)spk;
            }
        }
    }
}

__global__ __launch_bounds__(256) void spike_attn_kernel(
    const unsigned char* __restrict__ sq,
    const unsigned char* __restrict__ sk,
    const unsigned char* __restrict__ sv,
    float* __restrict__ outa)
{
    __shared__ unsigned char skLDS[32 * 64];
    __shared__ unsigned char svLDS[32 * 64];
    __shared__ _Float16 vT[64 * 32];
    __shared__ _Float16 attnT[8][16 * 32];

    const int bh   = blockIdx.y;
    const int bi   = bh / NHEADS;
    const int h    = bh % NHEADS;
    const int tid  = threadIdx.x;
    const int lane = tid & 31;
    const int wave = tid >> 5;
    const int hi   = lane >> 4;
    const int lm   = lane & 15;
    const int q0   = blockIdx.x * 128 + wave * 16;

    const unsigned base = (unsigned)bh * SEQ * HDIM;

    const unsigned char* qrow = sq + (size_t)base + (size_t)(q0 + lm) * HDIM;
    v8i aq;
    {
        v2i t0 = *(const v2i*)(qrow + 0  + 16 * hi);
        v2i t1 = *(const v2i*)(qrow + 8  + 16 * hi);
        v2i t2 = *(const v2i*)(qrow + 32 + 16 * hi);
        v2i t3 = *(const v2i*)(qrow + 40 + 16 * hi);
        aq[0] = t0[0]; aq[1] = t0[1]; aq[2] = t1[0]; aq[3] = t1[1];
        aq[4] = t2[0]; aq[5] = t2[1]; aq[6] = t3[0]; aq[7] = t3[1];
    }
    int q2h = 0;
#pragma unroll
    for (int i = 0; i < 8; ++i) q2h += __popc((unsigned)aq[i]);
    int q2full = q2h + __shfl_xor(q2h, 16, 32);
    float q2g[8];
#pragma unroll
    for (int g = 0; g < 8; ++g)
        q2g[g] = (float)__shfl(q2full, g + 8 * hi, 32);

    v8f oacc[4] = {};

    const unsigned skOff = lds_offset(skLDS);
    const unsigned svOff = lds_offset(svLDS);

    for (int key0 = 0; key0 < SEQ; key0 += 32) {
        __syncthreads();
        {
            int idx = (tid & 127) * 16;
            const unsigned char* gbase = (tid < 128) ? sk : sv;
            unsigned goff = base + (unsigned)key0 * HDIM + idx;
            unsigned loff = ((tid < 128) ? skOff : svOff) + idx;
            asm volatile("global_load_async_to_lds_b128 %0, %1, %2"
                         :: "v"(loff), "v"(goff), "s"(gbase) : "memory");
        }
        asm volatile("s_wait_asynccnt 0x0" ::: "memory");
        __syncthreads();

        {
            int m  = tid & 31;
            int d0 = (tid >> 5) * 8;
            unsigned long long raw =
                *(const unsigned long long*)(&svLDS[m * 64 + d0]);
#pragma unroll
            for (int i = 0; i < 8; ++i)
                vT[(d0 + i) * 32 + m] =
                    (_Float16)(float)((raw >> (8 * i)) & 0xffull);
        }
        __syncthreads();

#pragma unroll
        for (int sub = 0; sub < 2; ++sub) {
            const unsigned char* krow = &skLDS[(sub * 16 + lm) * 64];
            v4i u0 = *(const v4i*)(krow + 16 * hi);
            v4i u1 = *(const v4i*)(krow + 32 + 16 * hi);
            v8i bk_;
            bk_[0] = u0[0]; bk_[1] = u0[1]; bk_[2] = u0[2]; bk_[3] = u0[3];
            bk_[4] = u1[0]; bk_[5] = u1[1]; bk_[6] = u1[2]; bk_[7] = u1[3];

            int k2h = 0;
#pragma unroll
            for (int i = 0; i < 8; ++i) k2h += __popc((unsigned)bk_[i]);
            float k2f = (float)(k2h + __shfl_xor(k2h, 16, 32));

            v8i zero = {0, 0, 0, 0, 0, 0, 0, 0};
            v8i qk = __builtin_amdgcn_wmma_i32_16x16x64_iu8(
                false, aq, false, bk_, zero, false, false);

#pragma unroll
            for (int g = 0; g < 8; ++g) {
                float s = 64.0f - q2g[g] - k2f + 2.0f * (float)qk[g];
                float a = 1.0f / (1.0f + __expf(-s));
                attnT[wave][(g + 8 * hi) * 32 + sub * 16 + lm] = (_Float16)a;
            }
        }

        const _Float16* arow = &attnT[wave][lm * 32];
        v8h alo = *(const v8h*)(arow + 8 * hi);
        v8h ahi = *(const v8h*)(arow + 16 + 8 * hi);
        v16h aat;
#pragma unroll
        for (int i = 0; i < 8; ++i) { aat[i] = alo[i]; aat[8 + i] = ahi[i]; }

#pragma unroll
        for (int nc = 0; nc < 4; ++nc) {
            const _Float16* vrow = &vT[(nc * 16 + lm) * 32 + 8 * hi];
            v8h b0 = *(const v8h*)(vrow);
            v8h b1 = *(const v8h*)(vrow + 16);
            v16h bv_;
#pragma unroll
            for (int i = 0; i < 8; ++i) { bv_[i] = b0[i]; bv_[8 + i] = b1[i]; }
            oacc[nc] = __builtin_amdgcn_wmma_f32_16x16x32_f16(
                false, aat, false, bv_, (short)0, oacc[nc], false, false);
        }
    }

#pragma unroll
    for (int nc = 0; nc < 4; ++nc) {
        int d = nc * 16 + lm;
#pragma unroll
        for (int g = 0; g < 8; ++g) {
            int q = q0 + g + 8 * hi;
            outa[(size_t)(bi * SEQ + q) * DIMC + h * HDIM + d] = oacc[nc][g];
        }
    }
}

__global__ __launch_bounds__(256) void out_proj_kernel(
    const float* __restrict__ xin,
    const float* __restrict__ Wo,
    const float* __restrict__ bo,
    float* __restrict__ out)
{
    __shared__ float As[128 * 36];
    __shared__ float Bs[16 * 258];

    const int m0   = blockIdx.y * 128;
    const int n0   = blockIdx.x * 128;
    const int tid  = threadIdx.x;
    const int lane = tid & 31;
    const int wave = tid >> 5;
    const int mw   = wave >> 1;
    const int nw   = wave & 1;
    const int hi   = lane >> 4;
    const int lm   = lane & 15;

    v8f acc[2][4] = {};

    for (int k0 = 0; k0 < DIMC; k0 += 32) {
        __syncthreads();
        {
            int m  = tid >> 1;
            int kk = (tid & 1) * 16;
            const float* src = xin + (size_t)(m0 + m) * DIMC + k0 + kk;
#pragma unroll
            for (int i = 0; i < 16; i += 4)
                *(float4*)(&As[m * 36 + kk + i]) = *(const float4*)(src + i);
        }
        {
            int oc = tid >> 1;
            int kk = (tid & 1) * 16;
            const float* src = Wo + (size_t)(n0 + oc) * DIMC + k0 + kk;
#pragma unroll
            for (int i = 0; i < 16; i += 4) {
                float4 w4 = *(const float4*)(src + i);
                int k = kk + i;
                Bs[((k + 0) >> 1) * 258 + oc * 2 + ((k + 0) & 1)] = w4.x;
                Bs[((k + 1) >> 1) * 258 + oc * 2 + ((k + 1) & 1)] = w4.y;
                Bs[((k + 2) >> 1) * 258 + oc * 2 + ((k + 2) & 1)] = w4.z;
                Bs[((k + 3) >> 1) * 258 + oc * 2 + ((k + 3) & 1)] = w4.w;
            }
        }
        __syncthreads();

        const float* Arow0 = &As[(mw * 32 + lm) * 36];
        const float* Arow1 = &As[(mw * 32 + 16 + lm) * 36];
#pragma unroll
        for (int ks = 0; ks < 32; ks += 4) {
            v2f a0 = *(const v2f*)(&Arow0[ks + 2 * hi]);
            v2f a1 = *(const v2f*)(&Arow1[ks + 2 * hi]);
#pragma unroll
            for (int s = 0; s < 4; ++s) {
                int ocol = nw * 64 + s * 16 + lm;
                v2f b = *(const v2f*)(&Bs[((ks >> 1) + hi) * 258 + ocol * 2]);
                acc[0][s] = __builtin_amdgcn_wmma_f32_16x16x4_f32(
                    false, a0, false, b, (short)0, acc[0][s], false, false);
                acc[1][s] = __builtin_amdgcn_wmma_f32_16x16x4_f32(
                    false, a1, false, b, (short)0, acc[1][s], false, false);
            }
        }
    }

#pragma unroll
    for (int s = 0; s < 4; ++s) {
        int o  = n0 + nw * 64 + s * 16 + lm;
        float bb = bo[o];
#pragma unroll
        for (int ms = 0; ms < 2; ++ms) {
#pragma unroll
            for (int g = 0; g < 8; ++g) {
                int m = m0 + mw * 32 + ms * 16 + g + 8 * hi;
                out[(size_t)m * DIMC + o] = acc[ms][s][g] + bb;
            }
        }
    }
}

extern "C" void kernel_launch(void* const* d_in, const int* in_sizes, int n_in,
                              void* d_out, int out_size, void* d_ws,
                              size_t ws_size, hipStream_t stream)
{
    const float* x  = (const float*)d_in[0];
    const float* Wq = (const float*)d_in[1];
    const float* bq = (const float*)d_in[2];
    const float* Wk = (const float*)d_in[3];
    const float* bk = (const float*)d_in[4];
    const float* Wv = (const float*)d_in[5];
    const float* bv = (const float*)d_in[6];
    const float* Wo = (const float*)d_in[7];
    const float* bo = (const float*)d_in[8];

    const size_t spikeBytes = (size_t)BATCH * NHEADS * SEQ * HDIM;
    unsigned char* sq = (unsigned char*)d_ws;
    unsigned char* sk = sq + spikeBytes;
    unsigned char* sv = sk + spikeBytes;
    float* outa = (float*)(sv + spikeBytes);

    qkv_lif_kernel<<<dim3(2304 / 128, (BATCH * SEQ) / 128), 256, 0, stream>>>(
        x, Wq, Wk, Wv, bq, bk, bv, sq, sk, sv);

    spike_attn_kernel<<<dim3(SEQ / 128, BATCH * NHEADS), 256, 0, stream>>>(
        sq, sk, sv, outa);

    out_proj_kernel<<<dim3(DIMC / 128, (BATCH * SEQ) / 128), 256, 0, stream>>>(
        outa, Wo, bo, (float*)d_out);
}
